// MoDAttention_7687991460038
// MI455X (gfx1250) — compile-verified, matrix operand order checked by reading, not run on hardware
//
#include <hip/hip_runtime.h>
#include <hip/hip_bf16.h>
#include <stdint.h>

#define B_  4
#define L_  4096
#define D_  1024
#define H_  16
#define HD_ 64
#define K_  1024
#define NH_ 1024   // H*HD
#define SCALE_F 0.125f  // 1/sqrt(64)

typedef __attribute__((ext_vector_type(16))) __bf16 v16bf;
typedef __attribute__((ext_vector_type(8)))  float  v8f;
typedef __attribute__((ext_vector_type(4)))  unsigned int u32x4;

union FragU { v16bf v; u32x4 q[2]; };

__device__ __forceinline__ v16bf load_frag(const __bf16* p, int half) {
  // 16-bit A/B fragment: lanes 0-15 = row, half selects K-subgroup.
  // VGPRs 0-3: K = half*8 + 0..7 ; VGPRs 4-7: K = 16 + half*8 + 0..7
  FragU f;
  f.q[0] = *(const u32x4*)(p + half * 8);
  f.q[1] = *(const u32x4*)(p + half * 8 + 16);
  return f.v;
}

__device__ __forceinline__ v8f wmma_bf16(v16bf a, v16bf b, v8f c) {
  return __builtin_amdgcn_wmma_f32_16x16x32_bf16(false, a, false, b, (short)0, c,
                                                 false, false);
}

// ---------------- router scores ----------------
__global__ void mod_router_scores(const float* __restrict__ x,
                                  const float* __restrict__ w,
                                  const float* __restrict__ b,
                                  float* __restrict__ scores) {
  int t = blockIdx.x * blockDim.x + threadIdx.x;
  if (t >= B_ * L_) return;
  const float* xr = x + (size_t)t * D_;
  float acc = 0.f;
  for (int d = 0; d < D_; d += 4) {
    float4 xv = *(const float4*)(xr + d);
    float4 wv = *(const float4*)(w + d);
    acc += xv.x * wv.x + xv.y * wv.y + xv.z * wv.z + xv.w * wv.w;
  }
  scores[t] = acc + b[0];
}

// ---------------- per-batch bitonic top-k (full sort, descending) ----------------
__global__ void mod_topk_sort(const float* __restrict__ scores, int* __restrict__ idx) {
  __shared__ uint64_t keys[L_];  // 32 KB
  const int b = blockIdx.x;
  const int t = threadIdx.x;
  for (int i = t; i < L_; i += 1024) {
    uint32_t u = __float_as_uint(scores[b * L_ + i]);
    u = (u & 0x80000000u) ? ~u : (u | 0x80000000u);  // orderable float bits
    keys[i] = ((uint64_t)u << 32) | (uint32_t)(0xFFFFFFFFu - (uint32_t)i);
  }
  __syncthreads();
  for (int size = 2; size <= L_; size <<= 1) {
    for (int stride = size >> 1; stride > 0; stride >>= 1) {
      for (int p = t; p < L_ / 2; p += 1024) {
        int i = (p / stride) * (stride * 2) + (p % stride);
        int j = i + stride;
        bool desc = ((i & size) == 0);
        uint64_t a = keys[i], c = keys[j];
        if ((a < c) == desc) { keys[i] = c; keys[j] = a; }
      }
      __syncthreads();
    }
  }
  // first K entries: descending score, ties -> smaller index first
  idx[b * K_ + t] = (int)(0xFFFFFFFFu - (uint32_t)(keys[t] & 0xFFFFFFFFu));
}

// ---------------- weight convert + transpose (f32 [K,N] -> bf16 [N,K]) ----------------
__global__ void mod_convert_w(const float* __restrict__ w, __bf16* __restrict__ wt) {
  int t = blockIdx.x * 256 + threadIdx.x;   // 1024*1024 elements
  int r = t >> 10, c = t & 1023;
  wt[(size_t)c * 1024 + r] = (__bf16)w[t];
}

// ---------------- gather selected tokens to bf16 ----------------
__global__ void mod_gather(const float* __restrict__ x, const int* __restrict__ idx,
                           __bf16* __restrict__ xs) {
  int row = blockIdx.x;               // B*K rows
  int b = row >> 10, k = row & 1023;
  int tok = idx[b * K_ + k];
  const float* src = x + ((size_t)b * L_ + tok) * D_;
  __bf16* dst = xs + (size_t)row * D_;
  for (int d = threadIdx.x; d < D_; d += blockDim.x) dst[d] = (__bf16)src[d];
}

// ---------------- pass-through copy x -> out ----------------
__global__ void mod_copy_x(const float* __restrict__ x, float* __restrict__ out, int n4) {
  int t = blockIdx.x * blockDim.x + threadIdx.x;
  if (t < n4) ((float4*)out)[t] = ((const float4*)x)[t];
}

// ---------------- QKV GEMM: 2x2 register-blocked WMMA ----------------
// Each wave: 32x32 output; block of 8 waves: 128(M) x 64(N).
__global__ __launch_bounds__(256) void mod_qkv_gemm(
    const __bf16* __restrict__ xs, const __bf16* __restrict__ wqt,
    const __bf16* __restrict__ wkt, const __bf16* __restrict__ wvt,
    __bf16* __restrict__ qb, __bf16* __restrict__ kb, __bf16* __restrict__ vtb) {
  const int wave = threadIdx.x >> 5, lane = threadIdx.x & 31;
  const int lr = lane & 15, half = lane >> 4;
  const int mw = wave & 3, nw = wave >> 2;
  const int m0 = blockIdx.x * 128 + mw * 32;
  const int n0 = blockIdx.y * 64 + nw * 32;
  const int which = blockIdx.z;
  const __bf16* wt = (which == 0) ? wqt : ((which == 1) ? wkt : wvt);
  const __bf16* a0 = xs + (size_t)(m0 + lr) * D_;
  const __bf16* a1 = xs + (size_t)(m0 + 16 + lr) * D_;
  const __bf16* b0 = wt + (size_t)(n0 + lr) * D_;
  const __bf16* b1 = wt + (size_t)(n0 + 16 + lr) * D_;
  v8f c00 = {}, c01 = {}, c10 = {}, c11 = {};
#pragma unroll 2
  for (int kk = 0; kk < D_; kk += 32) {
    __builtin_prefetch(a0 + kk + 512, 0, 3);
    __builtin_prefetch(b0 + kk + 512, 0, 3);
    v16bf va0 = load_frag(a0 + kk, half);
    v16bf va1 = load_frag(a1 + kk, half);
    v16bf vb0 = load_frag(b0 + kk, half);
    v16bf vb1 = load_frag(b1 + kk, half);
    c00 = wmma_bf16(va0, vb0, c00);
    c01 = wmma_bf16(va0, vb1, c01);
    c10 = wmma_bf16(va1, vb0, c10);
    c11 = wmma_bf16(va1, vb1, c11);
  }
  // stores: batch / token base are wave-uniform (tiles never straddle a batch)
  const int bb = m0 >> 10;
  const int tokbase = (m0 & 1023) + 8 * half;
#pragma unroll
  for (int mt = 0; mt < 2; mt++) {
#pragma unroll
    for (int nt = 0; nt < 2; nt++) {
      const v8f& c = (mt == 0) ? ((nt == 0) ? c00 : c01) : ((nt == 0) ? c10 : c11);
      int col = n0 + nt * 16 + lr;
      int h = col >> 6, d = col & 63;
      size_t bh = (size_t)(bb * H_ + h);
      int tb = tokbase + mt * 16;
      if (which == 2) {
        // V transposed: consecutive tokens contiguous -> packed 32-bit stores
        __bf16* base = vtb + (bh * HD_ + d) * K_ + tb;
#pragma unroll
        for (int v = 0; v < 8; v += 2) {
          union { __bf16 h2[2]; unsigned int u; } pk;
          pk.h2[0] = (__bf16)c[v];
          pk.h2[1] = (__bf16)c[v + 1];
          *(unsigned int*)(base + v) = pk.u;
        }
      } else {
        __bf16* base = ((which == 0) ? qb : kb) + (bh * K_ + tb) * HD_ + d;
#pragma unroll
        for (int v = 0; v < 8; v++) base[(size_t)v * HD_] = (__bf16)c[v];
      }
    }
  }
}

// ---------------- causal flash attention over selected tokens ----------------
__global__ __launch_bounds__(256) void mod_attention(
    const __bf16* __restrict__ qb, const __bf16* __restrict__ kbuf,
    const __bf16* __restrict__ vtb, __bf16* __restrict__ oat) {
  __shared__ __attribute__((aligned(16))) __bf16 pshare[8 * 16 * 40];
  const int wave = threadIdx.x >> 5, lane = threadIdx.x & 31;
  const int lr = lane & 15, half = lane >> 4;
  const int bh = blockIdx.x;                       // b*16 + h
  const int q0 = blockIdx.y * 128 + wave * 16;
  const __bf16* qbase = qb + (size_t)bh * K_ * HD_;
  const __bf16* kbase = kbuf + (size_t)bh * K_ * HD_;
  const __bf16* vbase = vtb + (size_t)bh * HD_ * K_;
  __bf16* lds = pshare + wave * (16 * 40);

  const v16bf qa0 = load_frag(qbase + (size_t)(q0 + lr) * HD_, half);
  const v16bf qa1 = load_frag(qbase + (size_t)(q0 + lr) * HD_ + 32, half);

  float mrun[8], lrun[8];
#pragma unroll
  for (int v = 0; v < 8; v++) { mrun[v] = -1e30f; lrun[v] = 0.f; }
  v8f o0 = {}, o1 = {}, o2 = {}, o3 = {};

  const int nt = q0 / 32;                          // last 32-key tile (inclusive)
  for (int kt = 0; kt <= nt; kt++) {
    const int kb0 = kt * 32;
    v16bf kf0lo = load_frag(kbase + (size_t)(kb0 + lr) * HD_, half);
    v16bf kf0hi = load_frag(kbase + (size_t)(kb0 + lr) * HD_ + 32, half);
    v16bf kf1lo = load_frag(kbase + (size_t)(kb0 + 16 + lr) * HD_, half);
    v16bf kf1hi = load_frag(kbase + (size_t)(kb0 + 16 + lr) * HD_ + 32, half);
    v8f s0 = {}, s1 = {};
    s0 = wmma_bf16(qa0, kf0lo, s0);
    s0 = wmma_bf16(qa1, kf0hi, s0);
    s1 = wmma_bf16(qa0, kf1lo, s1);
    s1 = wmma_bf16(qa1, kf1hi, s1);
    const bool needmask = (kb0 + 31 > q0);
#pragma unroll
    for (int v = 0; v < 8; v++) {
      float a = s0[v] * SCALE_F;
      float bb = s1[v] * SCALE_F;
      if (needmask) {
        int qg = q0 + v + 8 * half;
        if (kb0 + lr > qg)      a = -1e30f;
        if (kb0 + 16 + lr > qg) bb = -1e30f;
      }
      float mt = fmaxf(a, bb);
#pragma unroll
      for (int off = 1; off < 16; off <<= 1) mt = fmaxf(mt, __shfl_xor(mt, off, 32));
      float mnew = fmaxf(mrun[v], mt);
      float corr = __expf(mrun[v] - mnew);
      float p0 = __expf(a - mnew);
      float p1 = __expf(bb - mnew);
      float rs = p0 + p1;
#pragma unroll
      for (int off = 1; off < 16; off <<= 1) rs += __shfl_xor(rs, off, 32);
      lrun[v] = lrun[v] * corr + rs;
      mrun[v] = mnew;
      o0[v] *= corr; o1[v] *= corr; o2[v] *= corr; o3[v] *= corr;
      int rowoff = (v + 8 * half) * 40;
      lds[rowoff + lr] = (__bf16)p0;
      lds[rowoff + 16 + lr] = (__bf16)p1;
    }
    // DS ops are in-order per wave; fence stops compiler reordering across the
    // cross-lane LDS exchange and drains the stores.
    asm volatile("s_wait_dscnt 0" ::: "memory");
    FragU pf;
    pf.q[0] = *(const u32x4*)(lds + lr * 40 + half * 8);
    pf.q[1] = *(const u32x4*)(lds + lr * 40 + 16 + half * 8);
    v16bf vf0 = load_frag(vbase + (size_t)(0 * 16 + lr) * K_ + kb0, half);
    v16bf vf1 = load_frag(vbase + (size_t)(1 * 16 + lr) * K_ + kb0, half);
    v16bf vf2 = load_frag(vbase + (size_t)(2 * 16 + lr) * K_ + kb0, half);
    v16bf vf3 = load_frag(vbase + (size_t)(3 * 16 + lr) * K_ + kb0, half);
    o0 = wmma_bf16(pf.v, vf0, o0);
    o1 = wmma_bf16(pf.v, vf1, o1);
    o2 = wmma_bf16(pf.v, vf2, o2);
    o3 = wmma_bf16(pf.v, vf3, o3);
    asm volatile("" ::: "memory");
  }

  const int bq = bh >> 4, h = bh & 15;
#pragma unroll
  for (int v = 0; v < 8; v++) {
    int tok = q0 + v + 8 * half;
    float inv = 1.f / lrun[v];
    size_t base = ((size_t)bq * K_ + tok) * NH_ + h * HD_ + lr;
    oat[base + 0]  = (__bf16)(o0[v] * inv);
    oat[base + 16] = (__bf16)(o1[v] * inv);
    oat[base + 32] = (__bf16)(o2[v] * inv);
    oat[base + 48] = (__bf16)(o3[v] * inv);
  }
}

// ---------------- output GEMM (2x2 blocked) + scatter into d_out ----------------
__global__ __launch_bounds__(256) void mod_out_gemm(
    const __bf16* __restrict__ oat, const __bf16* __restrict__ wot,
    const int* __restrict__ idx, float* __restrict__ out) {
  const int wave = threadIdx.x >> 5, lane = threadIdx.x & 31;
  const int lr = lane & 15, half = lane >> 4;
  const int mw = wave & 3, nw = wave >> 2;
  const int m0 = blockIdx.x * 128 + mw * 32;
  const int n0 = blockIdx.y * 64 + nw * 32;
  const __bf16* a0 = oat + (size_t)(m0 + lr) * NH_;
  const __bf16* a1 = oat + (size_t)(m0 + 16 + lr) * NH_;
  const __bf16* b0 = wot + (size_t)(n0 + lr) * NH_;
  const __bf16* b1 = wot + (size_t)(n0 + 16 + lr) * NH_;
  v8f c00 = {}, c01 = {}, c10 = {}, c11 = {};
#pragma unroll 2
  for (int kk = 0; kk < NH_; kk += 32) {
    __builtin_prefetch(a0 + kk + 512, 0, 3);
    __builtin_prefetch(b0 + kk + 512, 0, 3);
    v16bf va0 = load_frag(a0 + kk, half);
    v16bf va1 = load_frag(a1 + kk, half);
    v16bf vb0 = load_frag(b0 + kk, half);
    v16bf vb1 = load_frag(b1 + kk, half);
    c00 = wmma_bf16(va0, vb0, c00);
    c01 = wmma_bf16(va0, vb1, c01);
    c10 = wmma_bf16(va1, vb0, c10);
    c11 = wmma_bf16(va1, vb1, c11);
  }
  const int bb = m0 >> 10;
  const int ksbase = (m0 & 1023) + 8 * half;
  // resolve scatter targets once per M-tile (shared across both N-tiles)
  int tok0[8], tok1[8];
#pragma unroll
  for (int v = 0; v < 8; v++) {
    tok0[v] = idx[bb * K_ + ksbase + v];
    tok1[v] = idx[bb * K_ + ksbase + 16 + v];
  }
#pragma unroll
  for (int v = 0; v < 8; v++) {
    size_t r0 = ((size_t)bb * L_ + tok0[v]) * D_;
    size_t r1 = ((size_t)bb * L_ + tok1[v]) * D_;
    out[r0 + n0 + lr]      = c00[v];
    out[r0 + n0 + 16 + lr] = c01[v];
    out[r1 + n0 + lr]      = c10[v];
    out[r1 + n0 + 16 + lr] = c11[v];
  }
}

extern "C" void kernel_launch(void* const* d_in, const int* in_sizes, int n_in,
                              void* d_out, int out_size, void* d_ws, size_t ws_size,
                              hipStream_t stream) {
  (void)in_sizes; (void)n_in; (void)out_size; (void)ws_size;
  const float* x        = (const float*)d_in[0];
  const float* w_router = (const float*)d_in[1];
  const float* b_router = (const float*)d_in[2];
  const float* wq       = (const float*)d_in[3];
  const float* wk       = (const float*)d_in[4];
  const float* wv       = (const float*)d_in[5];
  const float* wo       = (const float*)d_in[6];
  float* out            = (float*)d_out;
  uint8_t* ws           = (uint8_t*)d_ws;

  // workspace layout (all 256B aligned)
  float*  scores = (float*)(ws + 0);                       // 65536 B
  int*    idx    = (int*)(ws + 65536);                     // 16384 B
  __bf16* xs     = (__bf16*)(ws + 81920);                  // 8 MB
  __bf16* wqt    = (__bf16*)(ws + 81920 + 8388608);        // 2 MB each
  __bf16* wkt    = (__bf16*)((uint8_t*)wqt + 2097152);
  __bf16* wvt    = (__bf16*)((uint8_t*)wkt + 2097152);
  __bf16* wot    = (__bf16*)((uint8_t*)wvt + 2097152);
  __bf16* qb     = (__bf16*)((uint8_t*)wot + 2097152);     // 8 MB
  __bf16* kb     = (__bf16*)((uint8_t*)qb + 8388608);      // 8 MB
  __bf16* vtb    = (__bf16*)((uint8_t*)kb + 8388608);      // 8 MB
  __bf16* oat    = (__bf16*)((uint8_t*)vtb + 8388608);     // 8 MB

  mod_router_scores<<<(B_ * L_ + 255) / 256, 256, 0, stream>>>(x, w_router, b_router, scores);
  mod_topk_sort<<<B_, 1024, 0, stream>>>(scores, idx);
  mod_convert_w<<<4096, 256, 0, stream>>>(wq, wqt);
  mod_convert_w<<<4096, 256, 0, stream>>>(wk, wkt);
  mod_convert_w<<<4096, 256, 0, stream>>>(wv, wvt);
  mod_convert_w<<<4096, 256, 0, stream>>>(wo, wot);
  mod_gather<<<B_ * K_, 256, 0, stream>>>(x, idx, xs);
  mod_copy_x<<<(B_ * L_ * D_ / 4 + 255) / 256, 256, 0, stream>>>(x, out, B_ * L_ * D_ / 4);
  mod_qkv_gemm<<<dim3(32, 16, 3), 256, 0, stream>>>(xs, wqt, wkt, wvt, qb, kb, vtb);
  mod_attention<<<dim3(B_ * H_, 8), 256, 0, stream>>>(qb, kb, vtb, oat);
  mod_out_gemm<<<dim3(32, 16), 256, 0, stream>>>(oat, wot, idx, out);
}
